// CrissCrossAttention_56633438765667
// MI455X (gfx1250) — matrix operand order repaired and checked by reading, not compiled, not run on hardware
//
#include <hip/hip_runtime.h>
#include <hip/hip_bf16.h>

// ---------------------------------------------------------------------------
// CrissCross attention for MI455X (gfx1250, wave32, WMMA 16x16x32 f16).
// LDS tiles live in WMMA *fragment order* (2x ds_load_b128 per fragment),
// GEMM uses double-buffered LDS with 64-wide K rounds (1 barrier / round).
// Staging keeps exactly 3 x uint4 in flight (no register spills).
// ---------------------------------------------------------------------------

typedef __attribute__((ext_vector_type(16))) _Float16 v16h;
typedef __attribute__((ext_vector_type(8)))  float    v8f;

#define BATCH  8
#define SEQ    4096
#define DIMENS 512
#define NHEADS 8
#define HD     64
#define GRID   64
#define MROWS  (BATCH * SEQ)   // 32768

union HF { v16h v; _Float16 h[16]; uint4 q[2]; };
union FA { v8f  v; float    f[8];  };

// Fragment-order inverse maps (CDNA5 ISA 7.12.2, 16-bit matrices):
//  A (16x32): element (m, K) -> lane = m + 16*((K>>3)&1), e = (K&7) + 8*(K>>4)
//             => 8-half chunk K=seg*8..+8 of row m is contiguous in one lane.
//  B (32x16): element (K, n) -> lane = 16*((K>>3)&1) + (n&15), e = (K&7) + 8*(K>>4)
//             => 8-half K-chunk of one column n is contiguous in one lane.

// ---------------------------------------------------------------------------
__global__ __launch_bounds__(256)
void cvt_f32_f16(const float* __restrict__ src, _Float16* __restrict__ dst, int n) {
  int i = blockIdx.x * blockDim.x + threadIdx.x;
  int stride = gridDim.x * blockDim.x;
  for (; i < n; i += stride) dst[i] = (_Float16)src[i];
}

// f32 src[R][C] -> f16 dst[C][R]   (transpose so GEMM-B is K-contiguous)
__global__ __launch_bounds__(256)
void cvt_transpose_f32_f16(const float* __restrict__ src, _Float16* __restrict__ dst,
                           int R, int C) {
  int i = blockIdx.x * blockDim.x + threadIdx.x;
  if (i < R * C) {
    int r = i / C, c = i % C;
    dst[(size_t)c * R + r] = (_Float16)src[i];
  }
}

// ---------------------------------------------------------------------------
// WMMA GEMM: C[M,N] = A[M,K] * Bt[N,K]^T   (A, Bt row-major f16)
// 256 threads = 8 waves; block tile 128x64; wave tile 32x32.
// Double-buffered LDS, 64-wide K rounds: stage(next) overlaps 8 WMMAs(cur).
// F16OUT=1: store f16.  F16OUT=0: store f32 + bias[n].
// ---------------------------------------------------------------------------
template <bool F16OUT>
__global__ __launch_bounds__(256)
void wmma_gemm(const _Float16* __restrict__ A, const _Float16* __restrict__ Bt,
               _Float16* __restrict__ Cf16, float* __restrict__ Cf32,
               const float* __restrict__ bias, int M, int N, int K) {
  // [buf][ks][slab/ct][lane][16]
  __shared__ __align__(16) _Float16 sAf[2][2][8][32][16];  // 2 x 16 KB
  __shared__ __align__(16) _Float16 sBf[2][2][4][32][16];  // 2 x  8 KB

  const int tid  = threadIdx.x;
  const int lane = tid & 31;
  const int wave = tid >> 5;
  const int wr   = wave >> 1;           // 0..3 -> row slabs 2*wr, 2*wr+1
  const int wc   = wave & 1;            // 0..1 -> col tiles 2*wc, 2*wc+1
  const int bm   = blockIdx.y * 128;
  const int bn   = blockIdx.x * 64;

  v8f z = {};
  v8f acc[2][2] = {{z, z}, {z, z}};

  // Stage one 64-wide K round into buffer `buf` (fragment order).
  // Two halves; each half keeps 3 uint4 (2 A-chunks + 1 B-chunk) in flight.
  auto stage = [&](int buf, int k0) {
#pragma unroll
    for (int half = 0; half < 2; ++half) {
      const int ccA0 = tid + (2 * half + 0) * 256;   // A: 1024 chunks total
      const int ccA1 = tid + (2 * half + 1) * 256;
      const int ccB  = tid + half * 256;             // B:  512 chunks total
      const int rA0 = ccA0 >> 3, sA0 = ccA0 & 7;
      const int rA1 = ccA1 >> 3, sA1 = ccA1 & 7;
      const int nB  = ccB  >> 3, sB  = ccB  & 7;

      uint4 da0 = *(const uint4*)(A  + (size_t)(bm + rA0) * K + k0 + sA0 * 8);
      uint4 da1 = *(const uint4*)(A  + (size_t)(bm + rA1) * K + k0 + sA1 * 8);
      uint4 db  = *(const uint4*)(Bt + (size_t)(bn + nB)  * K + k0 + sB  * 8);

      { int ks = sA0 >> 2, sr = sA0 & 3;
        *(uint4*)&sAf[buf][ks][rA0 >> 4][(rA0 & 15) + ((sr & 1) << 4)][(sr >> 1) << 3] = da0; }
      { int ks = sA1 >> 2, sr = sA1 & 3;
        *(uint4*)&sAf[buf][ks][rA1 >> 4][(rA1 & 15) + ((sr & 1) << 4)][(sr >> 1) << 3] = da1; }
      { int ks = sB >> 2, sr = sB & 3;
        *(uint4*)&sBf[buf][ks][nB >> 4][((sr & 1) << 4) + (nB & 15)][(sr >> 1) << 3] = db; }
    }
  };

  stage(0, 0);
  int buf = 0;
  for (int k0 = 0; k0 < K; k0 += 64) {
    __syncthreads();                            // buffer `buf` is ready
    if (k0 + 64 < K) stage(buf ^ 1, k0 + 64);   // overlap next round's staging
    if (k0 + 128 < K) {                         // global_prefetch_b8 round+2
      __builtin_prefetch(A  + (size_t)(bm + (tid >> 1)) * K + k0 + 128, 0, 1);
      __builtin_prefetch(Bt + (size_t)(bn + (tid >> 2)) * K + k0 + 128, 0, 1);
    }
#pragma unroll
    for (int ks = 0; ks < 2; ++ks) {
      HF a0, a1, b0, b1;
      a0.q[0] = *(const uint4*)&sAf[buf][ks][wr * 2 + 0][lane][0];
      a0.q[1] = *(const uint4*)&sAf[buf][ks][wr * 2 + 0][lane][8];
      a1.q[0] = *(const uint4*)&sAf[buf][ks][wr * 2 + 1][lane][0];
      a1.q[1] = *(const uint4*)&sAf[buf][ks][wr * 2 + 1][lane][8];
      b0.q[0] = *(const uint4*)&sBf[buf][ks][wc * 2 + 0][lane][0];
      b0.q[1] = *(const uint4*)&sBf[buf][ks][wc * 2 + 0][lane][8];
      b1.q[0] = *(const uint4*)&sBf[buf][ks][wc * 2 + 1][lane][0];
      b1.q[1] = *(const uint4*)&sBf[buf][ks][wc * 2 + 1][lane][8];

      acc[0][0] = __builtin_amdgcn_wmma_f32_16x16x32_f16(false, a0.v, false, b0.v,
                                                         (short)0, acc[0][0], false, false);
      acc[0][1] = __builtin_amdgcn_wmma_f32_16x16x32_f16(false, a0.v, false, b1.v,
                                                         (short)0, acc[0][1], false, false);
      acc[1][0] = __builtin_amdgcn_wmma_f32_16x16x32_f16(false, a1.v, false, b0.v,
                                                         (short)0, acc[1][0], false, false);
      acc[1][1] = __builtin_amdgcn_wmma_f32_16x16x32_f16(false, a1.v, false, b1.v,
                                                         (short)0, acc[1][1], false, false);
    }
    buf ^= 1;
  }

  // C/D layout: VGPR r -> row r (lanes 0-15) / 8+r (lanes 16-31), col = lane&15
#pragma unroll
  for (int si = 0; si < 2; ++si) {
#pragma unroll
    for (int ti = 0; ti < 2; ++ti) {
      const int mrow = bm + (wr * 2 + si) * 16 + ((lane >> 4) << 3);
      const int nc   = bn + (wc * 2 + ti) * 16 + (lane & 15);
      FA r; r.v = acc[si][ti];
#pragma unroll
      for (int k = 0; k < 8; ++k) {
        size_t idx = (size_t)(mrow + k) * N + nc;
        if (F16OUT) Cf16[idx] = (_Float16)r.f[k];
        else        Cf32[idx] = r.f[k] + bias[nc];
      }
    }
  }
}

// ---------------------------------------------------------------------------
// Criss-cross attention, one block per (b, head, line). block = 128 (4 waves).
// MODE 0: row attention (keys stride 1), store.  MODE 1: column (stride G), add.
// ---------------------------------------------------------------------------
template <int MODE>
__global__ __launch_bounds__(128)
void cc_attn(const _Float16* __restrict__ Qm, const _Float16* __restrict__ Km,
             const _Float16* __restrict__ Vm, _Float16* __restrict__ Om) {
  __shared__ __align__(16) _Float16 sQf[2][4][32][16];  // Q (A-frag), reused for P
  __shared__ __align__(16) _Float16 sKf[2][4][32][16];  // K (B-frag for Q*K^T)
  __shared__ __align__(16) _Float16 sVf[2][4][32][16];  // V (B-frag for P*V)
  __shared__ float sS[64][65];                          // scores (pad -> no conflicts)

  const int tid  = threadIdx.x;
  const int lane = tid & 31;
  const int wave = tid >> 5;

  const int bid  = blockIdx.x;
  const int b    = bid / (NHEADS * GRID);
  const int rem  = bid % (NHEADS * GRID);
  const int h    = rem / GRID;
  const int line = rem % GRID;
  const size_t base = (size_t)b * SEQ + (MODE == 0 ? (size_t)line * GRID : (size_t)line);
  const int colbase = h * HD;

  // ---- stage Q, K, V: per iteration, 3 loads in flight then 3 stores ------
#pragma unroll
  for (int it = 0; it < 4; ++it) {
    int cc  = tid + it * 128;
    int row = cc >> 3;              // query/key index within the line
    int seg = cc & 7;               // 8-half chunk of d
    size_t flat = base + (MODE == 0 ? (size_t)row : (size_t)row * GRID);
    size_t off  = flat * DIMENS + colbase + seg * 8;
    uint4 dq = *(const uint4*)(Qm + off);
    uint4 dk = *(const uint4*)(Km + off);
    uint4 dv = *(const uint4*)(Vm + off);

    int ks = seg >> 2, sr = seg & 3;
    // Q -> A-fragment order (m = query, K-dim = d)
    *(uint4*)&sQf[ks][row >> 4][(row & 15) + ((sr & 1) << 4)][(sr >> 1) << 3] = dq;
    // K -> B-fragment order (n = key, K-dim = d): ct = key>>4
    *(uint4*)&sKf[ks][row >> 4][((sr & 1) << 4) + (row & 15)][(sr >> 1) << 3] = dk;
    // V -> B-fragment order (n = d, K-dim = key): 8-way lane scatter, e fixed
    HF tv; tv.q[0] = dv;
    int vks = row >> 5, kr = row & 31;
    int kg = kr >> 4, e = kr & 15, ct = seg >> 1, d0 = (seg & 1) << 3;
#pragma unroll
    for (int j = 0; j < 8; ++j)
      sVf[vks][ct][(kg << 4) + d0 + j][e] = tv.h[j];
  }
  __syncthreads();

  const float scale = 0.125f;  // 1/sqrt(64)

  // -------- S = Q * K^T  (wave: query rows [16w,16w+16), 4 key tiles) ------
  v8f z = {};
  v8f acc[4] = {z, z, z, z};
#pragma unroll
  for (int ks = 0; ks < 2; ++ks) {
    HF aq;
    aq.q[0] = *(const uint4*)&sQf[ks][wave][lane][0];
    aq.q[1] = *(const uint4*)&sQf[ks][wave][lane][8];
#pragma unroll
    for (int tt = 0; tt < 4; ++tt) {
      HF bk;
      bk.q[0] = *(const uint4*)&sKf[ks][tt][lane][0];
      bk.q[1] = *(const uint4*)&sKf[ks][tt][lane][8];
      acc[tt] = __builtin_amdgcn_wmma_f32_16x16x32_f16(false, aq.v, false, bk.v,
                                                       (short)0, acc[tt], false, false);
    }
  }
#pragma unroll
  for (int tt = 0; tt < 4; ++tt) {
    FA r; r.v = acc[tt];
#pragma unroll
    for (int k = 0; k < 8; ++k)
      sS[wave * 16 + ((lane >> 4) << 3) + k][tt * 16 + (lane & 15)] = r.f[k] * scale;
  }
  __syncthreads();

  // -------- softmax over 64 keys; P written in A-fragment order into sQf ---
  if (tid < 64) {
    const int q = tid;
    float mx = -1e30f;
    for (int j = 0; j < 64; ++j) mx = fmaxf(mx, sS[q][j]);
    float sum = 0.f;
    for (int j = 0; j < 64; ++j) { float e = __expf(sS[q][j] - mx); sS[q][j] = e; sum += e; }
    const float inv = 1.0f / sum;
    const int wr = q >> 4, m = q & 15;
#pragma unroll
    for (int g = 0; g < 8; ++g) {     // keys j = g*8 .. g*8+7 -> one lane, contiguous
      HF u;
#pragma unroll
      for (int j = 0; j < 8; ++j) u.h[j] = (_Float16)(sS[q][g * 8 + j] * inv);
      int ks = g >> 2, gr = g & 3;
      *(uint4*)&sQf[ks][wr][m + ((gr >> 1) << 4)][(gr & 1) << 3] = u.q[0];
    }
  }
  __syncthreads();

  // -------- O = P * V ------------------------------------------------------
  v8f out[4] = {z, z, z, z};
#pragma unroll
  for (int ks = 0; ks < 2; ++ks) {
    HF ap;
    ap.q[0] = *(const uint4*)&sQf[ks][wave][lane][0];
    ap.q[1] = *(const uint4*)&sQf[ks][wave][lane][8];
#pragma unroll
    for (int tt = 0; tt < 4; ++tt) {
      HF bv;
      bv.q[0] = *(const uint4*)&sVf[ks][tt][lane][0];
      bv.q[1] = *(const uint4*)&sVf[ks][tt][lane][8];
      out[tt] = __builtin_amdgcn_wmma_f32_16x16x32_f16(false, ap.v, false, bv.v,
                                                       (short)0, out[tt], false, false);
    }
  }
#pragma unroll
  for (int tt = 0; tt < 4; ++tt) {
    FA r; r.v = out[tt];
#pragma unroll
    for (int k = 0; k < 8; ++k) {
      int qm = wave * 16 + ((lane >> 4) << 3) + k;
      size_t flat = base + (MODE == 0 ? (size_t)qm : (size_t)qm * GRID);
      size_t idx  = flat * DIMENS + colbase + tt * 16 + (lane & 15);
      if (MODE == 0) Om[idx] = (_Float16)r.f[k];
      else           Om[idx] = (_Float16)((float)Om[idx] + r.f[k]);
    }
  }
}

// ---------------------------------------------------------------------------
extern "C" void kernel_launch(void* const* d_in, const int* in_sizes, int n_in,
                              void* d_out, int out_size, void* d_ws, size_t ws_size,
                              hipStream_t stream) {
  (void)in_sizes; (void)n_in; (void)out_size; (void)ws_size;

  const float* x  = (const float*)d_in[0];
  const float* Wq = (const float*)d_in[1];
  const float* Wk = (const float*)d_in[2];
  const float* Wv = (const float*)d_in[3];
  const float* Wo = (const float*)d_in[4];
  const float* bo = (const float*)d_in[5];
  float* out = (float*)d_out;

  const size_t MD = (size_t)MROWS * DIMENS;   // 16,777,216 halves
  const size_t WW = (size_t)DIMENS * DIMENS;  //    262,144 halves

  _Float16* xh  = (_Float16*)d_ws;  // reused as attention output after QKV
  _Float16* qh  = xh + MD;
  _Float16* kh  = qh + MD;
  _Float16* vh  = kh + MD;
  _Float16* wqt = vh + MD;          // transposed f16 weights [N][K]
  _Float16* wkt = wqt + WW;
  _Float16* wvt = wkt + WW;
  _Float16* wot = wvt + WW;

  // 1) f32 -> f16 conversions (weights transposed for K-contiguous B staging)
  cvt_f32_f16<<<8192, 256, 0, stream>>>(x, xh, (int)MD);
  cvt_transpose_f32_f16<<<1024, 256, 0, stream>>>(Wq, wqt, DIMENS, DIMENS);
  cvt_transpose_f32_f16<<<1024, 256, 0, stream>>>(Wk, wkt, DIMENS, DIMENS);
  cvt_transpose_f32_f16<<<1024, 256, 0, stream>>>(Wv, wvt, DIMENS, DIMENS);
  cvt_transpose_f32_f16<<<1024, 256, 0, stream>>>(Wo, wot, DIMENS, DIMENS);

  // 2) Q/K/V projections (WMMA)
  dim3 ggrid(DIMENS / 64, MROWS / 128);  // (8, 256)
  wmma_gemm<true><<<ggrid, 256, 0, stream>>>(xh, wqt, qh, nullptr, nullptr,
                                             MROWS, DIMENS, DIMENS);
  wmma_gemm<true><<<ggrid, 256, 0, stream>>>(xh, wkt, kh, nullptr, nullptr,
                                             MROWS, DIMENS, DIMENS);
  wmma_gemm<true><<<ggrid, 256, 0, stream>>>(xh, wvt, vh, nullptr, nullptr,
                                             MROWS, DIMENS, DIMENS);

  // 3) row + column attention; x buffer is dead -> reuse as attn output
  const int nblk = BATCH * NHEADS * GRID;  // 4096
  cc_attn<0><<<nblk, 128, 0, stream>>>(qh, kh, vh, xh);
  cc_attn<1><<<nblk, 128, 0, stream>>>(qh, kh, vh, xh);

  // 4) output projection + bias (WMMA, f32 out)
  wmma_gemm<false><<<ggrid, 256, 0, stream>>>(xh, wot, nullptr, out, bo,
                                              MROWS, DIMENS, DIMENS);
}
